// TopKAttentionPooling_80805514707216
// MI455X (gfx1250) — compile-verified, matrix operand order checked by reading, not run on hardware
//
#include <hip/hip_runtime.h>
#include <hip/hip_bf16.h>
#include <stdint.h>

// ---------------------------------------------------------------------------
// TopKAttentionPooling on MI455X (gfx1250, wave32)
//   Stage 1: scores = ReLU(x@W1+b1)@W2+b2 via v_wmma_f32_16x16x32_bf16
//            (memory-bound: 410 MB x-stream @23.3TB/s ~ 17.6us; bf16 WMMA
//             keeps the 26 GFLOP GEMM under the HBM roofline)
//   Stage 2: 4-pass radix select (top-k threshold) + index selection
//   Stage 3: gather 256 rows (1 MB) + mean -> [1024]
// ---------------------------------------------------------------------------

typedef __attribute__((ext_vector_type(16))) __bf16 v16bf;
typedef __attribute__((ext_vector_type(8)))  float  v8f;

#define DIM 1024
#define HID 128
#define JT 8              // HID / 16 column tiles
#define SLICES 4          // K sliced into 4 LDS-resident chunks
#define KK_PER_SLICE 8    // 8 k-steps of 32 per slice  (4*8*32 = 1024 = DIM)
#define BLOCK_THREADS 512
#define WAVES 16          // 512 / 32

// --------------------------------------------------------------------------
// Pre-pack W1 (f32, DIMxHID) into bf16 B-fragments, layout [kk][j][lane][e]
// so the GEMM loop reads 32 contiguous bytes per lane (2x b128).
// Fragment element mapping (16-bit B matrix 32x16, ISA 7.12.2):
//   lane = {g=lane>>4, n=lane&15},  e = 2*v+h
//   k_local = (e & 7) + g*8 + (e>=8 ? 16 : 0)
// --------------------------------------------------------------------------
__global__ void prepack_w1(const float* __restrict__ W1,
                           __bf16* __restrict__ packed) {
  int t = blockIdx.x * blockDim.x + threadIdx.x;      // 32*8*32*16 = 131072
  int e    = t & 15;
  int lane = (t >> 4) & 31;
  int j    = (t >> 9) & 7;
  int kk   = (t >> 12) & 31;
  int g = lane >> 4, n = lane & 15;
  int k = kk * 32 + (e & 7) + g * 8 + ((e >= 8) ? 16 : 0);
  packed[t] = (__bf16)W1[k * HID + j * 16 + n];       // native v_cvt, RNE
}

// --------------------------------------------------------------------------
// Scoring GEMM: each wave owns a 16-row tile, keeps 16x128 f32 accumulators
// in registers (8x v8f), workgroup stages 64KB W1 slices in LDS.
// --------------------------------------------------------------------------
__global__ __launch_bounds__(BLOCK_THREADS)
void score_kernel(const float* __restrict__ x,
                  const __bf16* __restrict__ w1p,
                  const float* __restrict__ b1,
                  const float* __restrict__ W2,
                  const float* __restrict__ b2,
                  float* __restrict__ scores,
                  int nTiles, int nRounds) {
  __shared__ uint4 lds[4096];                        // 64 KB W1 slice
  const int tid  = threadIdx.x;
  const int wave = tid >> 5;
  const int lane = tid & 31;
  const int g = lane >> 4;
  const int n = lane & 15;

  float w2v[JT], b1v[JT];
#pragma unroll
  for (int j = 0; j < JT; ++j) {                     // broadcast loads (L2 hot)
    w2v[j] = W2[j * 16 + n];
    b1v[j] = b1[j * 16 + n];
  }
  const float b2v = b2[0];
  const int tilesPerRound = gridDim.x * WAVES;

  for (int round = 0; round < nRounds; ++round) {
    const int tile = round * tilesPerRound + blockIdx.x * WAVES + wave;
    const bool active = tile < nTiles;

    v8f acc[JT];
    const v8f vzero = {0.f, 0.f, 0.f, 0.f, 0.f, 0.f, 0.f, 0.f};
#pragma unroll
    for (int j = 0; j < JT; ++j) acc[j] = vzero;

    const float* xrow = x + (size_t)(active ? tile : 0) * 16 * DIM;

    for (int s = 0; s < SLICES; ++s) {
      __syncthreads();
      { // cooperative 64KB slice copy, fully coalesced b128s
        const uint4* src = (const uint4*)w1p + s * 4096;
#pragma unroll
        for (int i = 0; i < 4096 / BLOCK_THREADS; ++i)
          lds[tid + i * BLOCK_THREADS] = src[tid + i * BLOCK_THREADS];
      }
      __syncthreads();

      if (active) {
#pragma unroll
        for (int kl = 0; kl < KK_PER_SLICE; ++kl) {
          const int kk = s * KK_PER_SLICE + kl;
          // ---- A fragment: row m = n, two contiguous 8-float runs ----
          union { float4 f4[4]; float f[16]; } au;
          const float4* p =
              (const float4*)(xrow + (size_t)n * DIM + kk * 32 + g * 8);
          au.f4[0] = p[0]; au.f4[1] = p[1];          // k_local 0..7  (+g*8)
          au.f4[2] = p[4]; au.f4[3] = p[5];          // k_local 16..23(+g*8)
          v16bf av;
#pragma unroll
          for (int e = 0; e < 16; ++e) av[e] = (__bf16)au.f[e];  // hw cvt

          // ---- 8 column tiles: B from LDS, WMMA accumulate ----
#pragma unroll
          for (int j = 0; j < JT; ++j) {
            union { uint4 u[2]; v16bf v; } bf;
            const uint4* bp = &lds[((kl * JT + j) * 32 + lane) * 2];
            bf.u[0] = bp[0]; bf.u[1] = bp[1];
            acc[j] = __builtin_amdgcn_wmma_f32_16x16x32_bf16(
                false, av, false, bf.v, (short)0, acc[j], false, false);
          }
        }
      }
    }

    if (active) {
      // scores[m] = b2 + sum_n ReLU(h[m][n] + b1[n]) * W2[n]
      float part[8];
#pragma unroll
      for (int r = 0; r < 8; ++r) {
        float ssum = 0.f;
#pragma unroll
        for (int j = 0; j < JT; ++j)
          ssum += fmaxf(acc[j][r] + b1v[j], 0.f) * w2v[j];
        part[r] = ssum;
      }
#pragma unroll
      for (int mask = 1; mask < 16; mask <<= 1) {    // 16-lane butterfly
#pragma unroll
        for (int r = 0; r < 8; ++r)
          part[r] += __shfl_xor(part[r], mask, 32);
      }
      if (n == 0) {                                  // lane 0 / 16 write m=g*8+r
        float4 o0 = {part[0] + b2v, part[1] + b2v, part[2] + b2v, part[3] + b2v};
        float4 o1 = {part[4] + b2v, part[5] + b2v, part[6] + b2v, part[7] + b2v};
        float* sp = scores + (size_t)tile * 16 + g * 8;
        *(float4*)(sp)     = o0;
        *(float4*)(sp + 4) = o1;
      }
    }
  }
}

// --------------------------------------------------------------------------
// Top-k: radix select over sortable keys (larger float -> larger uint)
// --------------------------------------------------------------------------
__device__ __forceinline__ unsigned sortKey(float f) {
  unsigned u = __float_as_uint(f);
  return (u & 0x80000000u) ? ~u : (u | 0x80000000u);
}

__global__ void topk_hist(const float* __restrict__ scores, int n,
                          const unsigned* __restrict__ state,
                          unsigned maskHi, int shift,
                          unsigned* __restrict__ hist) {
  __shared__ unsigned h[256];
  if (threadIdx.x < 256) h[threadIdx.x] = 0;
  __syncthreads();
  const unsigned prefix = state[0] & maskHi;
  for (int i = blockIdx.x * blockDim.x + threadIdx.x; i < n;
       i += gridDim.x * blockDim.x) {
    unsigned u = sortKey(scores[i]);
    if ((u & maskHi) == prefix) atomicAdd(&h[(u >> shift) & 255u], 1u);
  }
  __syncthreads();
  if (threadIdx.x < 256 && h[threadIdx.x])
    atomicAdd(&hist[threadIdx.x], h[threadIdx.x]);
}

__global__ void topk_scan(unsigned* hist, unsigned* state, int shift,
                          const int* __restrict__ dK) {
  if (threadIdx.x == 0) {
    unsigned kneed = (shift == 24) ? (unsigned)(*dK) : state[1];
    unsigned cum = 0;
    int b = 255;
    for (; b > 0; --b) {
      unsigned c = hist[b];
      if (cum + c >= kneed) break;
      cum += c;
    }
    state[0] |= ((unsigned)b) << shift;
    state[1] = kneed - cum;
  }
  __syncthreads();
  if (threadIdx.x < 256) hist[threadIdx.x] = 0;      // ready for next pass
}

__global__ void topk_select(const float* __restrict__ scores, int n,
                            const unsigned* __restrict__ state,
                            const int* __restrict__ dK,
                            int* __restrict__ idxOut,
                            unsigned* __restrict__ cnt) {
  const unsigned T = state[0];
  const unsigned kneed = state[1];
  const unsigned K = (unsigned)(*dK);
  const unsigned G = K - kneed;                      // strictly-greater count
  for (int i = blockIdx.x * blockDim.x + threadIdx.x; i < n;
       i += gridDim.x * blockDim.x) {
    unsigned u = sortKey(scores[i]);
    if (u > T) {
      unsigned p = atomicAdd(&cnt[0], 1u);
      if (p < G) idxOut[p] = i;
    } else if (u == T) {
      unsigned p = atomicAdd(&cnt[1], 1u);
      if (p < kneed) idxOut[G + p] = i;
    }
  }
}

// --------------------------------------------------------------------------
// Mean of selected rows: 8 partial blocks (deterministic) + reduce
// --------------------------------------------------------------------------
__global__ __launch_bounds__(1024)
void gather_partial(const float* __restrict__ x, const int* __restrict__ idx,
                    const int* __restrict__ dK, float* __restrict__ partial) {
  const int K = *dK;
  const int rpb = K / 8;
  const int d = threadIdx.x;
  const int r0 = blockIdx.x * rpb;
  const int r1 = (blockIdx.x == 7) ? K : r0 + rpb;
  float ssum = 0.f;
  for (int r = r0; r < r1; ++r)
    ssum += x[(size_t)idx[r] * DIM + d];
  partial[blockIdx.x * DIM + d] = ssum;
}

__global__ __launch_bounds__(1024)
void mean_final(const float* __restrict__ partial, const int* __restrict__ dK,
                float* __restrict__ out) {
  const int d = threadIdx.x;
  float ssum = 0.f;
#pragma unroll
  for (int b = 0; b < 8; ++b) ssum += partial[b * DIM + d];
  out[d] = ssum / (float)(*dK);
}

// --------------------------------------------------------------------------
extern "C" void kernel_launch(void* const* d_in, const int* in_sizes, int n_in,
                              void* d_out, int out_size, void* d_ws,
                              size_t ws_size, hipStream_t stream) {
  const float* x  = (const float*)d_in[0];
  const float* W1 = (const float*)d_in[1];
  const float* b1 = (const float*)d_in[2];
  const float* W2 = (const float*)d_in[3];
  const float* b2 = (const float*)d_in[4];
  const int*   dK = (const int*)d_in[5];
  float* out = (float*)d_out;

  const int N = in_sizes[0] / DIM;
  const int nTiles = N / 16;                         // N=100000 -> 6250 exact

  // ---- workspace layout ----
  char* ws = (char*)d_ws;
  __bf16* w1p = (__bf16*)(ws);                              // 256 KB packed W1
  float* scores = (float*)(ws + 262144);                    // N floats
  size_t off = 262144 + (size_t)N * 4;
  off = (off + 255) & ~(size_t)255;
  unsigned* hist  = (unsigned*)(ws + off);                  // 1024 B
  unsigned* state = (unsigned*)(ws + off + 1024);           // {prefix, kneed}
  unsigned* cnt   = (unsigned*)(ws + off + 1032);           // {cntG, cntE}
  int*      idx   = (int*)(ws + off + 1040);                // up to 1024 ints
  float* partial  = (float*)(ws + off + 1040 + 4096);       // 8*DIM floats

  hipMemsetAsync(hist, 0, 1024 + 8 + 8, stream);           // hist+state+cnt

  prepack_w1<<<512, 256, 0, stream>>>(W1, w1p);

  const int grid = 256;
  const int tilesPerRound = grid * WAVES;
  const int nRounds = (nTiles + tilesPerRound - 1) / tilesPerRound;
  score_kernel<<<grid, BLOCK_THREADS, 0, stream>>>(x, w1p, b1, W2, b2, scores,
                                                   nTiles, nRounds);

  const int      shifts[4] = {24, 16, 8, 0};
  const unsigned masks[4]  = {0u, 0xFF000000u, 0xFFFF0000u, 0xFFFFFF00u};
  for (int p = 0; p < 4; ++p) {
    topk_hist<<<64, 256, 0, stream>>>(scores, N, state, masks[p], shifts[p], hist);
    topk_scan<<<1, 256, 0, stream>>>(hist, state, shifts[p], dK);
  }
  topk_select<<<64, 256, 0, stream>>>(scores, N, state, dK, idx, cnt);

  gather_partial<<<8, 1024, 0, stream>>>(x, idx, dK, partial);
  mean_final<<<1, DIM, 0, stream>>>(partial, dK, out);
}
